// ST_Encoder_Module_18442589569459
// MI455X (gfx1250) — matrix operand order repaired and checked by reading, not compiled, not run on hardware
//
#include <hip/hip_runtime.h>
#include <hip/hip_bf16.h>
#include <math.h>

// ---------------- constants matching the reference ----------------
#define N_NODE   40000
#define N_REL    2000
#define N_TRIPLE 600000
#define DIM      128
#define F3       384   // DIM*(DEPTH+1)
#define NPROXY   64
#define EPSN     1e-12f

typedef __attribute__((ext_vector_type(16))) _Float16 v16h;
typedef __attribute__((ext_vector_type(8)))  _Float16 v8h;
typedef __attribute__((ext_vector_type(8)))  float    v8f;

// ---------------- small helpers ----------------
__device__ __forceinline__ float wave_sum(float v) {
  #pragma unroll
  for (int off = 16; off; off >>= 1) v += __shfl_xor(v, off, 32);
  return v;
}
__device__ __forceinline__ float wave_max(float v) {
  #pragma unroll
  for (int off = 16; off; off >>= 1) v = fmaxf(v, __shfl_xor(v, off, 32));
  return v;
}
// order-preserving float -> u32 key (for atomic max on signed floats)
__device__ __forceinline__ unsigned fkey(float f) {
  unsigned u = __float_as_uint(f);
  return (u & 0x80000000u) ? ~u : (u | 0x80000000u);
}
__device__ __forceinline__ float fdec(unsigned k) {
  unsigned u = (k & 0x80000000u) ? (k ^ 0x80000000u) : ~k;
  return __uint_as_float(u);
}

// ---------------- fill kernels ----------------
__global__ void fill_f32_k(float* p, float v, long n) {
  long i = (long)blockIdx.x * blockDim.x + threadIdx.x;
  if (i < n) p[i] = v;
}
__global__ void fill_u32_k(unsigned* p, unsigned v, long n) {
  long i = (long)blockIdx.x * blockDim.x + threadIdx.x;
  if (i < n) p[i] = v;
}

// ---------------- _avg: mean over neighbors ----------------
// one wave per edge: acc[row] += emb[col]; deg[row]++
__global__ __launch_bounds__(256) void avg_scatter_k(
    const int* __restrict__ row, const int* __restrict__ col,
    const float* __restrict__ emb, float* __restrict__ acc, int* __restrict__ deg, int T) {
  int wave = (int)(((long)blockIdx.x * blockDim.x + threadIdx.x) >> 5);
  int lane = threadIdx.x & 31;
  if (wave >= T) return;
  int r = row[wave], c = col[wave];
  float4 v = ((const float4*)(emb + (size_t)c * DIM))[lane];
  float* dst = acc + (size_t)r * DIM + lane * 4;
  atomicAdd(dst + 0, v.x); atomicAdd(dst + 1, v.y);
  atomicAdd(dst + 2, v.z); atomicAdd(dst + 3, v.w);
  if (lane == 0) atomicAdd(&deg[r], 1);
}
__global__ void avg_div_k(float* acc, const int* deg, long n) {
  long i = (long)blockIdx.x * blockDim.x + threadIdx.x;
  if (i >= n) return;
  int d = deg[i >> 7];
  acc[i] = d ? acc[i] / (float)d : 0.f;
}

// ---------------- row-wise L2 normalize (wave per row) ----------------
__global__ __launch_bounds__(256) void norm_rows_k(
    const float* __restrict__ X, float* __restrict__ Y, int D, int nrows) {
  int row = (int)(((long)blockIdx.x * blockDim.x + threadIdx.x) >> 5);
  int lane = threadIdx.x & 31;
  if (row >= nrows) return;
  const float* x = X + (size_t)row * D;
  float ss = 0.f;
  for (int c = lane; c < D; c += 32) { float v = x[c]; ss += v * v; }
  ss = wave_sum(ss);
  float inv = 1.f / fmaxf(sqrtf(ss), EPSN);
  float* y = Y + (size_t)row * D;
  for (int c = lane; c < D; c += 32) y[c] = x[c] * inv;
}
// only the 1/norm (for nout = l2norm(out) applied lazily in the GEMM)
__global__ __launch_bounds__(256) void row_inv_norm_k(
    const float* __restrict__ X, int ld, float* __restrict__ rowinv, int D, int nrows) {
  int row = (int)(((long)blockIdx.x * blockDim.x + threadIdx.x) >> 5);
  int lane = threadIdx.x & 31;
  if (row >= nrows) return;
  const float* x = X + (size_t)row * ld;
  float ss = 0.f;
  for (int c = lane; c < D; c += 32) { float v = x[c]; ss += v * v; }
  ss = wave_sum(ss);
  if (lane == 0) rowinv[row] = 1.f / fmaxf(sqrtf(ss), EPSN);
}

// ---------------- per-relation attention logit ----------------
__global__ __launch_bounds__(256) void rel_logit_k(
    const float* __restrict__ rel_norm, const float* __restrict__ attn,
    float* __restrict__ logit, int R) {
  int row = (int)(((long)blockIdx.x * blockDim.x + threadIdx.x) >> 5);
  int lane = threadIdx.x & 31;
  if (row >= R) return;
  const float* x = rel_norm + (size_t)row * DIM;
  float d = 0.f;
  #pragma unroll
  for (int c = lane; c < DIM; c += 32) d += x[c] * attn[c];
  d = wave_sum(d);
  if (lane == 0) logit[row] = d;
}

// ---------------- sparse softmax passes (thread per edge) ----------------
__global__ void edge_max_k(const int* __restrict__ row, const int* __restrict__ rid,
                           const float* __restrict__ logit, unsigned* __restrict__ mkey, int T) {
  int t = (int)((long)blockIdx.x * blockDim.x + threadIdx.x);
  if (t >= T) return;
  atomicMax(&mkey[row[t]], fkey(logit[rid[t]]));
}
__global__ void edge_expsum_k(const int* __restrict__ row, const int* __restrict__ rid,
                              const float* __restrict__ logit, const unsigned* __restrict__ mkey,
                              float* __restrict__ e_edge, float* __restrict__ s_node, int T) {
  int t = (int)((long)blockIdx.x * blockDim.x + threadIdx.x);
  if (t >= T) return;
  int r = row[t];
  float e = __expf(logit[rid[t]] - fdec(mkey[r]));
  e_edge[t] = e;
  atomicAdd(&s_node[r], e);
}

// ---------------- main edge kernel: Householder reflection + weighted scatter ----
// one wave per edge; feats/rel_norm are L2-resident (20.5 MB / 1 MB)
__global__ __launch_bounds__(256) void edge_scatter_k(
    const int* __restrict__ row, const int* __restrict__ col, const int* __restrict__ rid,
    const float* __restrict__ e_edge, const float* __restrict__ s_node,
    const float* __restrict__ feats, const float* __restrict__ rel_norm,
    float* __restrict__ acc, int T) {
  int wave = (int)(((long)blockIdx.x * blockDim.x + threadIdx.x) >> 5);
  int lane = threadIdx.x & 31;
  if (wave >= T) return;
  int r = row[wave], c = col[wave], q = rid[wave];
  float4 a  = ((const float4*)(feats    + (size_t)c * DIM))[lane];
  float4 rv = ((const float4*)(rel_norm + (size_t)q * DIM))[lane];
  float d = a.x * rv.x + a.y * rv.y + a.z * rv.z + a.w * rv.w;
  d = wave_sum(d);
  float w = e_edge[wave] / s_node[r];
  float s = 2.f * d;
  float* dst = acc + (size_t)r * DIM + lane * 4;
  atomicAdd(dst + 0, (a.x - s * rv.x) * w);
  atomicAdd(dst + 1, (a.y - s * rv.y) * w);
  atomicAdd(dst + 2, (a.z - s * rv.z) * w);
  atomicAdd(dst + 3, (a.w - s * rv.w) * w);
}

// ---------------- tanh + write into concat slab ----------------
__global__ void tanh_store_k(const float* __restrict__ src, float* __restrict__ feats,
                             float* __restrict__ slab, int ldo, long n) {
  long i = (long)blockIdx.x * blockDim.x + threadIdx.x;
  if (i >= n) return;
  float t = tanhf(src[i]);
  feats[i] = t;
  long rrow = i >> 7;
  slab[rrow * ldo + (i & 127)] = t;
}

// ---------------- row softmax over 64 (wave per row) ----------------
__global__ __launch_bounds__(256) void softmax64_k(float* __restrict__ pa, int nrows) {
  int row = (int)(((long)blockIdx.x * blockDim.x + threadIdx.x) >> 5);
  int lane = threadIdx.x & 31;
  if (row >= nrows) return;
  float* p = pa + (size_t)row * NPROXY;
  float a = p[lane], b = p[lane + 32];
  float m = wave_max(fmaxf(a, b));
  float ea = __expf(a - m), eb = __expf(b - m);
  float s = wave_sum(ea + eb);
  p[lane] = ea / s; p[lane + 32] = eb / s;
}

// ---------------- WMMA GEMM (f16 in, f32 acc), block tile 128x64, 8 waves ----
// Double-buffered, fragment-ordered LDS: each lane's v16h fragment is one
// contiguous 32-byte ds read; staging is global_load_b128 -> ds_store_b128.
// Next tile's global loads issue before the current tile's WMMAs (overlap),
// one barrier per K-step. All per-thread staging state is K-loop invariant.
//
// A 16x32 fragment layout: lane=(r&15)+16*bit3(c), half=(c&7)+8*bit4(c)
// B 32x16 fragment layout: lane=(n&15)+16*bit3(k), half=(k&7)+8*bit4(k)
//
// mode 1: pa     = (A*rowscale) @ B^T                        (proxy matching logits)
// mode 2: pf     = out - A @ B                               (pa @ proxy subtract)
// mode 3: d_out  = g*out + (1-g)*pf,  g = sigmoid(A@B + bias) (gating)
#define BM 128
#define BK 32
#define BN 64
__global__ __launch_bounds__(256) void gemm_wmma_k(
    const float* __restrict__ A, int lda, const float* __restrict__ rowscale,
    const float* __restrict__ B, int ldb, int transB,
    int M, int K, int mode,
    float* __restrict__ pa, int ldc,
    const float* __restrict__ outp, int ldo,
    float* __restrict__ pf,
    const float* __restrict__ bias,
    float* __restrict__ finalp, int ldf) {
  __shared__ __align__(32) _Float16 As_frag[2][BM * BK];  // 2 x 8 KB
  __shared__ __align__(32) _Float16 Bs_frag[2][BK * BN];  // 2 x 4 KB
  const int tid = threadIdx.x;
  const int wid = tid >> 5, lane = tid & 31;
  const int row0 = blockIdx.x * BM;
  const int col0 = blockIdx.y * BN;

  // ---- K-loop-invariant per-thread staging state ----
  // A: thread owns chunks (r_lo, cc) and (r_lo+64, cc), 8 contiguous k each
  const int r_lo = tid >> 2, ccA = tid & 3;
  const int gr0 = row0 + r_lo, gr1 = row0 + r_lo + 64;
  const bool ok0 = gr0 < M, ok1 = gr1 < M;
  const float sc0 = (ok0 && rowscale) ? rowscale[gr0] : 1.f;
  const float sc1 = (ok1 && rowscale) ? rowscale[gr1] : 1.f;
  const float* pA0 = A + (size_t)(ok0 ? gr0 : 0) * lda + ccA * 8;
  const float* pA1 = A + (size_t)(ok1 ? gr1 : 0) * lda + ccA * 8;
  const int aAddr0 = (((r_lo >> 4) * 32) + (r_lo & 15) + ((ccA & 1) << 4)) * 16 + ((ccA >> 1) << 3);
  const int r_hi = r_lo + 64;
  const int aAddr1 = (((r_hi >> 4) * 32) + (r_hi & 15) + ((ccA & 1) << 4)) * 16 + ((ccA >> 1) << 3);
  // B: thread owns chunk (n, kc), 8 contiguous k
  const int nB = tid & 63, kcB = tid >> 6;
  const float* pBt = B + (size_t)(col0 + nB) * ldb + kcB * 8;        // transB source
  const float* pBn = B + (size_t)(kcB * 8) * ldb + col0 + nB;        // normal source
  const int bAddr = (((nB >> 4) * 32) + (nB & 15) + ((kcB & 1) << 4)) * 16 + ((kcB >> 1) << 3);

  auto cvt8 = [](float4 p0, float4 p1, float sc) {
    v8h hv;
    hv[0] = (_Float16)(p0.x * sc); hv[1] = (_Float16)(p0.y * sc);
    hv[2] = (_Float16)(p0.z * sc); hv[3] = (_Float16)(p0.w * sc);
    hv[4] = (_Float16)(p1.x * sc); hv[5] = (_Float16)(p1.y * sc);
    hv[6] = (_Float16)(p1.z * sc); hv[7] = (_Float16)(p1.w * sc);
    return hv;
  };
  auto stage = [&](int buf, int k0) {
    v8h z; 
    #pragma unroll
    for (int j = 0; j < 8; ++j) z[j] = (_Float16)0.f;
    // A chunk 0
    v8h a0 = z, a1 = z;
    if (ok0) {
      const float4* s = (const float4*)(pA0 + k0);
      a0 = cvt8(s[0], s[1], sc0);
    }
    if (ok1) {
      const float4* s = (const float4*)(pA1 + k0);
      a1 = cvt8(s[0], s[1], sc1);
    }
    *(v8h*)&As_frag[buf][aAddr0] = a0;
    *(v8h*)&As_frag[buf][aAddr1] = a1;
    // B chunk
    v8h bv;
    if (transB) {
      const float4* s = (const float4*)(pBt + k0);
      bv = cvt8(s[0], s[1], 1.f);
    } else {
      const float* s = pBn + (size_t)k0 * ldb;
      #pragma unroll
      for (int j = 0; j < 8; ++j) bv[j] = (_Float16)s[(size_t)j * ldb];
    }
    *(v8h*)&Bs_frag[buf][bAddr] = bv;
  };

  v8f acc[4];
  #pragma unroll
  for (int g = 0; g < 4; ++g)
    #pragma unroll
    for (int j = 0; j < 8; ++j) acc[g][j] = 0.f;

  const int nk = K / BK;
  stage(0, 0);
  __syncthreads();

  const int fragA = (wid * 32 + lane) * 16;
  for (int kt = 0; kt < nk; ++kt) {
    const int cur = kt & 1;
    // load all fragments for this tile (wide 32B ds reads, distinct registers)
    v16h af  = *(const v16h*)&As_frag[cur][fragA];
    v16h bf0 = *(const v16h*)&Bs_frag[cur][(0 * 32 + lane) * 16];
    v16h bf1 = *(const v16h*)&Bs_frag[cur][(1 * 32 + lane) * 16];
    v16h bf2 = *(const v16h*)&Bs_frag[cur][(2 * 32 + lane) * 16];
    v16h bf3 = *(const v16h*)&Bs_frag[cur][(3 * 32 + lane) * 16];
    // overlap: issue next tile's global loads + LDS stores before the WMMAs
    if (kt + 1 < nk) stage(cur ^ 1, (kt + 1) * BK);
    // back-to-back matrix ops
    acc[0] = __builtin_amdgcn_wmma_f32_16x16x32_f16(false, af, false, bf0, (short)0, acc[0], false, false);
    acc[1] = __builtin_amdgcn_wmma_f32_16x16x32_f16(false, af, false, bf1, (short)0, acc[1], false, false);
    acc[2] = __builtin_amdgcn_wmma_f32_16x16x32_f16(false, af, false, bf2, (short)0, acc[2], false, false);
    acc[3] = __builtin_amdgcn_wmma_f32_16x16x32_f16(false, af, false, bf3, (short)0, acc[3], false, false);
    __syncthreads();
  }

  // C/D layout: lane L -> n = L%16; VGPR j -> m = j + (L/16)*8
  const int nlane = lane & 15;
  const int mhi   = (lane >> 4) << 3;
  #pragma unroll
  for (int g = 0; g < 4; ++g) {
    int n = col0 + (g << 4) + nlane;
    #pragma unroll
    for (int j = 0; j < 8; ++j) {
      int m = row0 + (wid << 4) + mhi + j;
      if (m >= M) continue;
      float v = acc[g][j];
      if (mode == 1) {
        pa[(size_t)m * ldc + n] = v;
      } else if (mode == 2) {
        pf[(size_t)m * F3 + n] = outp[(size_t)m * ldo + n] - v;
      } else {
        float z = v + bias[n];
        float gg = 1.f / (1.f + __expf(-z));
        float o = outp[(size_t)m * ldo + n];
        float p = pf[(size_t)m * F3 + n];
        finalp[(size_t)m * ldf + n] = gg * o + (1.f - gg) * p;
      }
    }
  }
}

// ---------------- host orchestration ----------------
static inline int blks(long n, int t) { return (int)((n + t - 1) / t); }

extern "C" void kernel_launch(void* const* d_in, const int* in_sizes, int n_in,
                              void* d_out, int out_size, void* d_ws, size_t ws_size,
                              hipStream_t stream) {
  const int*   ent_adj = (const int*)d_in[0];
  const int*   rel_adj = (const int*)d_in[1];
  const int*   adj     = (const int*)d_in[4];
  const int*   r_index = (const int*)d_in[5];
  // r_val (d_in[6]) cancels: tri_rel = l2norm(r_val * rel_emb[rid]) = l2norm(rel_emb[rid])
  const float* ent_emb = (const float*)d_in[9];
  const float* rel_emb = (const float*)d_in[10];
  const float* e_attn  = (const float*)d_in[11];
  const float* e_gate  = (const float*)d_in[12];
  const float* e_proxy = (const float*)d_in[13];
  const float* e_bias  = (const float*)d_in[14];
  const float* r_attn  = (const float*)d_in[15];
  const float* r_gate  = (const float*)d_in[16];
  const float* r_proxy = (const float*)d_in[17];
  const float* r_bias  = (const float*)d_in[18];
  float* out = (float*)d_out;  // [N, 768]
  (void)in_sizes; (void)n_in; (void)out_size; (void)ws_size;

  // bump-allocate workspace (reused by both sequential encoders), ~137 MB
  char* ws = (char*)d_ws;
  size_t off = 0;
  auto carve = [&](size_t bytes) -> void* {
    void* p = ws + off;
    off = (off + bytes + 255) & ~(size_t)255;
    return p;
  };
  float*    feature   = (float*)carve((size_t)N_NODE * DIM * 4);
  float*    feats     = (float*)carve((size_t)N_NODE * DIM * 4);
  float*    accum     = (float*)carve((size_t)N_NODE * DIM * 4);
  float*    pf        = (float*)carve((size_t)N_NODE * F3 * 4);
  float*    pa        = (float*)carve((size_t)N_NODE * NPROXY * 4);
  float*    e_edge    = (float*)carve((size_t)N_TRIPLE * 4);
  float*    s_node    = (float*)carve((size_t)N_NODE * 4);
  unsigned* mkey      = (unsigned*)carve((size_t)N_NODE * 4);
  float*    rowinv    = (float*)carve((size_t)N_NODE * 4);
  float*    rel_norm  = (float*)carve((size_t)N_REL * DIM * 4);
  float*    rel_logit = (float*)carve((size_t)N_REL * 4);
  float*    nproxy    = (float*)carve((size_t)NPROXY * F3 * 4);
  int*      deg       = (int*)carve((size_t)N_NODE * 4);

  const int* adj_row = adj;
  const int* adj_col = adj + N_TRIPLE;
  const int* rid     = r_index + N_TRIPLE;

  const long nNF = (long)N_NODE * DIM;
  const int edge_waves_blocks = blks((long)N_TRIPLE * 32, 256);

  // normalized relation embeddings (shared by both encoders)
  norm_rows_k<<<blks((long)N_REL * 32, 256), 256, 0, stream>>>(rel_emb, rel_norm, DIM, N_REL);

  auto run_encoder = [&](const int* avg_row, const int* avg_col, const float* emb,
                         const float* attn, const float* gatew, const float* proxy,
                         const float* bias, int encOff) {
    float* oslab = out + encOff;  // [N,384] slice of d_out, ld 768

    // ---- _avg: mean over neighbors ----
    fill_f32_k<<<blks(nNF, 256), 256, 0, stream>>>(feature, 0.f, nNF);
    fill_u32_k<<<blks(N_NODE, 256), 256, 0, stream>>>((unsigned*)deg, 0u, N_NODE);
    avg_scatter_k<<<edge_waves_blocks, 256, 0, stream>>>(avg_row, avg_col, emb, feature, deg, N_TRIPLE);
    avg_div_k<<<blks(nNF, 256), 256, 0, stream>>>(feature, deg, nNF);

    // ---- NR encoder ----
    tanh_store_k<<<blks(nNF, 256), 256, 0, stream>>>(feature, feats, oslab, 768, nNF);
    for (int l = 0; l < 2; ++l) {
      rel_logit_k<<<blks((long)N_REL * 32, 256), 256, 0, stream>>>(rel_norm, attn + l * DIM, rel_logit, N_REL);
      fill_u32_k<<<blks(N_NODE, 256), 256, 0, stream>>>(mkey, 0u, N_NODE);
      fill_f32_k<<<blks(N_NODE, 256), 256, 0, stream>>>(s_node, 0.f, N_NODE);
      edge_max_k<<<blks(N_TRIPLE, 256), 256, 0, stream>>>(adj_row, rid, rel_logit, mkey, N_TRIPLE);
      edge_expsum_k<<<blks(N_TRIPLE, 256), 256, 0, stream>>>(adj_row, rid, rel_logit, mkey, e_edge, s_node, N_TRIPLE);
      fill_f32_k<<<blks(nNF, 256), 256, 0, stream>>>(accum, 0.f, nNF);
      edge_scatter_k<<<edge_waves_blocks, 256, 0, stream>>>(adj_row, adj_col, rid, e_edge, s_node,
                                                            feats, rel_norm, accum, N_TRIPLE);
      tanh_store_k<<<blks(nNF, 256), 256, 0, stream>>>(accum, feats, oslab + (l + 1) * DIM, 768, nNF);
    }

    // ---- proxy matching + gating (WMMA GEMMs) ----
    row_inv_norm_k<<<blks((long)N_NODE * 32, 256), 256, 0, stream>>>(oslab, 768, rowinv, F3, N_NODE);
    norm_rows_k<<<blks((long)NPROXY * 32, 256), 256, 0, stream>>>(proxy, nproxy, F3, NPROXY);

    dim3 g1(blks(N_NODE, BM), 1);
    gemm_wmma_k<<<g1, 256, 0, stream>>>(oslab, 768, rowinv, nproxy, F3, 1,
                                        N_NODE, F3, 1, pa, NPROXY,
                                        nullptr, 0, nullptr, nullptr, nullptr, 0);
    softmax64_k<<<blks((long)N_NODE * 32, 256), 256, 0, stream>>>(pa, N_NODE);

    dim3 g2(blks(N_NODE, BM), F3 / BN);
    gemm_wmma_k<<<g2, 256, 0, stream>>>(pa, NPROXY, nullptr, proxy, F3, 0,
                                        N_NODE, NPROXY, 2, nullptr, 0,
                                        oslab, 768, pf, nullptr, nullptr, 0);

    gemm_wmma_k<<<g2, 256, 0, stream>>>(pf, F3, nullptr, gatew, F3, 0,
                                        N_NODE, F3, 3, nullptr, 0,
                                        oslab, 768, pf, bias, oslab, 768);
  };

  // encoder over entity graph -> out[:, 0:384]
  run_encoder(ent_adj, ent_adj + N_TRIPLE, ent_emb, e_attn, e_gate, e_proxy, e_bias, 0);
  // encoder over relation graph -> out[:, 384:768]
  run_encoder(rel_adj, rel_adj + N_TRIPLE, rel_emb, r_attn, r_gate, r_proxy, r_bias, F3);
}
